// GATVGAEEncoder_16810501996995
// MI455X (gfx1250) — matrix operand order repaired and checked by reading, not compiled, not run on hardware
//
#include <hip/hip_runtime.h>

typedef __attribute__((ext_vector_type(16))) _Float16 v16h;
typedef __attribute__((ext_vector_type(8)))  float    v8f;

#define NNODES 50000
#define NEDGE  800000
#define ETOT   (NEDGE + NNODES)
#define INC    128          // K for every GEMM in this model
#define H1C    4
#define H2C    2
#define CHID   32
#define NEG_SLOPE 0.2f
#define MWAVES 624          // M-tile waves per N-tile column (grid-stride)

// ---------------------------------------------------------------------------
// WMMA GEMM, K fixed at 128: C[M,Nc] = A[M,128] @ B[128,Nc]
// f32 in/out, f16 multiply, f32 accumulate.  M%16==0, Nc%16==0.
// One wave32 owns one 16-column strip: it loads the 128x16 B panel once into
// registers (4 x v16h fragments), then grid-strides over 16-row A tiles,
// issuing 4 fully-unrolled v_wmma_f32_16x16x32_f16 per tile.
// Fragment layouts used by this code (l16 = lane&15, half = lane>>4; A/B use contiguous K):
//   A 16x32: row M=l16; e -> K=16*half+e
//   B 32x16: col N=l16; e -> K=16*half+e
//   D 16x16 f32: vgpr r -> row M=r+8*half, col N=l16
// ---------------------------------------------------------------------------
__global__ void wmma_gemm128_kernel(const float* __restrict__ A,
                                    const float* __restrict__ B,
                                    float* __restrict__ C, int M, int Nc) {
  int wave = (int)((blockIdx.x * blockDim.x + threadIdx.x) >> 5);
  int lane = (int)(threadIdx.x & 31);
  int ntN  = Nc >> 4;
  int tN   = wave % ntN;            // N-tile (uniform per wave)
  int mw   = wave / ntN;            // starting M-tile index
  int l16  = lane & 15;
  int half = lane >> 4;
  int col  = tN * 16 + l16;

  // Load the 128x16 B panel for this column strip once (f32 -> f16).
  v16h bfrag[4];
#pragma unroll
  for (int ks = 0; ks < 4; ++ks) {
    int kb = ks * 32 + 16 * half;
#pragma unroll
    for (int e = 0; e < 16; ++e)
      bfrag[ks][e] = (_Float16)B[(size_t)(kb + e) * Nc + col];
  }

  for (int tM = mw; tM * 16 < M; tM += MWAVES) {   // wave-uniform loop
    const float* Arow = A + (size_t)(tM * 16 + l16) * INC;
    v8f acc = {};
#pragma unroll
    for (int ks = 0; ks < 4; ++ks) {
      v16h a;
      int ka = ks * 32 + 16 * half;
#pragma unroll
      for (int e = 0; e < 8; ++e) {
        a[e]     = (_Float16)Arow[ka + e];
        a[e + 8] = (_Float16)Arow[ka + 8 + e];
      }
      acc = __builtin_amdgcn_wmma_f32_16x16x32_f16(false, a, false, bfrag[ks],
                                                   (short)0, acc, false, false);
    }
    int rbase = tM * 16 + 8 * half;
#pragma unroll
    for (int r = 0; r < 8; ++r)
      C[(size_t)(rbase + r) * Nc + col] = acc[r];
  }
}

// --------------------------- attention logits ------------------------------
__global__ void attn_logits_kernel(const float* __restrict__ xl,
                                   const float* __restrict__ a_src,
                                   const float* __restrict__ a_dst,
                                   float* __restrict__ alS, float* __restrict__ alD,
                                   int n, int H, int C) {
  int t = blockIdx.x * blockDim.x + threadIdx.x;
  if (t >= n * H) return;
  int node = t / H, h = t - node * H;
  const float* xp = xl + ((size_t)node * H + h) * C;
  float s = 0.f, d = 0.f;
  for (int c = 0; c < C; ++c) {
    float v = xp[c];
    s += v * a_src[h * C + c];
    d += v * a_dst[h * C + c];
  }
  alS[t] = s;
  alD[t] = d;
}

// monotone order-preserving float<->uint encoding (for atomic max; init = 0)
__device__ __forceinline__ unsigned enc_f(float f) {
  unsigned u = __float_as_uint(f);
  return (u & 0x80000000u) ? ~u : (u | 0x80000000u);
}
__device__ __forceinline__ float dec_f(unsigned u) {
  return (u & 0x80000000u) ? __uint_as_float(u ^ 0x80000000u)
                           : __uint_as_float(~u);
}

// ----------------- edge pass 1: leaky-relu score + segment max -------------
__global__ void edge_score_max_kernel(const int* __restrict__ srcE,
                                      const int* __restrict__ dstE,
                                      const float* __restrict__ alS,
                                      const float* __restrict__ alD,
                                      float* __restrict__ esc,
                                      unsigned* __restrict__ menc, int H) {
  int t = blockIdx.x * blockDim.x + threadIdx.x;
  if (t >= ETOT * H) return;
  int e = t / H, h = t - e * H;
  int s, d;
  if (e < NEDGE) { s = srcE[e]; d = dstE[e]; } else { s = d = e - NEDGE; }
  float v = alS[s * H + h] + alD[d * H + h];
  v = v > 0.f ? v : NEG_SLOPE * v;
  esc[t] = v;
  atomicMax(&menc[d * H + h], enc_f(v));
}

// ----------------- edge pass 2: exp(e - m) + segment sum -------------------
__global__ void edge_expsum_kernel(const int* __restrict__ dstE,
                                   float* __restrict__ esc,
                                   const unsigned* __restrict__ menc,
                                   float* __restrict__ den, int H) {
  int t = blockIdx.x * blockDim.x + threadIdx.x;
  if (t >= ETOT * H) return;
  int e = t / H, h = t - e * H;
  int d = (e < NEDGE) ? dstE[e] : e - NEDGE;
  float ex = expf(esc[t] - dec_f(menc[d * H + h]));
  esc[t] = ex;
  atomicAdd(&den[d * H + h], ex);
}

// ---------- edge pass 3: acc[dst,h,c] += alpha * xl[src,h,c], c = lane -----
__global__ void edge_aggregate_kernel(const int* __restrict__ srcE,
                                      const int* __restrict__ dstE,
                                      const float* __restrict__ xl,
                                      const float* __restrict__ esc,
                                      const float* __restrict__ den,
                                      float* __restrict__ acc, int H) {
  int t = blockIdx.x * blockDim.x + threadIdx.x;   // < ETOT*H*32
  if (t >= ETOT * H * CHID) return;
  int c  = t & (CHID - 1);
  int eh = t >> 5;                                  // CHID == 32
  int e  = eh / H, h = eh - e * H;
  int s, d;
  if (e < NEDGE) { s = srcE[e]; d = dstE[e]; } else { s = d = e - NEDGE; }
  float alpha = esc[eh] / (den[d * H + h] + 1e-16f);
  atomicAdd(&acc[((size_t)d * H + h) * CHID + c],
            alpha * xl[((size_t)s * H + h) * CHID + c]);
}

// ------------------------- finalize: bias + ELU ----------------------------
__global__ void finalize_elu_kernel(const float* __restrict__ acc,
                                    const float* __restrict__ bias,
                                    float* __restrict__ out, int n, int F) {
  int t = blockIdx.x * blockDim.x + threadIdx.x;
  if (t >= n * F) return;
  float v = acc[t] + bias[t % F];
  out[t] = v > 0.f ? v : (expf(v) - 1.f);
}

// -------------------- finalize: head-mean + bias ---------------------------
__global__ void finalize_mean_kernel(const float* __restrict__ acc,
                                     const float* __restrict__ bias,
                                     float* __restrict__ out, int n, int H, int C) {
  int t = blockIdx.x * blockDim.x + threadIdx.x;
  if (t >= n * C) return;
  int node = t / C, c = t - node * C;
  float v = 0.f;
  for (int h = 0; h < H; ++h) v += acc[((size_t)node * H + h) * C + c];
  out[t] = v / (float)H + bias[c];
}

// ---------------------------------------------------------------------------
static void run_gat(const float* xin, const float* W, const float* a_s,
                    const float* a_d, const float* bias, int H, bool concat,
                    const int* srcE, const int* dstE,
                    float* xl, float* acc, float* esc, float* alS, float* alD,
                    unsigned* menc, float* den, float* out, hipStream_t stream) {
  const int n = NNODES;
  const int F = H * CHID;

  // xl = xin @ W   [n, F]; K = 128.  ntN * MWAVES total waves, 8 waves/block.
  int ntN = F / 16;
  int blocks = (ntN * MWAVES) / 8;      // MWAVES=624 keeps this integral for ntN 4/8
  wmma_gemm128_kernel<<<blocks, 256, 0, stream>>>(xin, W, xl, n, F);

  attn_logits_kernel<<<(n * H + 255) / 256, 256, 0, stream>>>(xl, a_s, a_d,
                                                              alS, alD, n, H, CHID);

  hipMemsetAsync(menc, 0, (size_t)n * H * sizeof(unsigned), stream);
  hipMemsetAsync(den, 0, (size_t)n * H * sizeof(float), stream);
  hipMemsetAsync(acc, 0, (size_t)n * F * sizeof(float), stream);

  int eh = ETOT * H;
  edge_score_max_kernel<<<(eh + 255) / 256, 256, 0, stream>>>(srcE, dstE, alS, alD,
                                                              esc, menc, H);
  edge_expsum_kernel<<<(eh + 255) / 256, 256, 0, stream>>>(dstE, esc, menc, den, H);
  int ehc = eh * CHID;
  edge_aggregate_kernel<<<(ehc + 255) / 256, 256, 0, stream>>>(srcE, dstE, xl, esc,
                                                               den, acc, H);
  if (concat)
    finalize_elu_kernel<<<(n * F + 255) / 256, 256, 0, stream>>>(acc, bias, out, n, F);
  else
    finalize_mean_kernel<<<(n * CHID + 255) / 256, 256, 0, stream>>>(acc, bias, out,
                                                                     n, H, CHID);
}

extern "C" void kernel_launch(void* const* d_in, const int* in_sizes, int n_in,
                              void* d_out, int out_size, void* d_ws, size_t ws_size,
                              hipStream_t stream) {
  const float* x    = (const float*)d_in[0];
  const int*   ei   = (const int*)d_in[1];     // edge_index [2, E]
  const float* W1   = (const float*)d_in[2];
  const float* a1s  = (const float*)d_in[3];
  const float* a1d  = (const float*)d_in[4];
  const float* b1   = (const float*)d_in[5];
  const float* Wmu  = (const float*)d_in[6];
  const float* amus = (const float*)d_in[7];
  const float* amud = (const float*)d_in[8];
  const float* bmu  = (const float*)d_in[9];
  const float* Wls  = (const float*)d_in[10];
  const float* alss = (const float*)d_in[11];
  const float* alsd = (const float*)d_in[12];
  const float* bls  = (const float*)d_in[13];

  const int* srcE = ei;
  const int* dstE = ei + NEDGE;

  // workspace carving (sizes are all multiples of 256 bytes)
  char* ws = (char*)d_ws;
  size_t off = 0;
  float* xl  = (float*)(ws + off); off += (size_t)NNODES * INC * 4;   // 25.6 MB
  float* h1  = (float*)(ws + off); off += (size_t)NNODES * INC * 4;   // 25.6 MB
  float* acc = (float*)(ws + off); off += (size_t)NNODES * INC * 4;   // 25.6 MB
  float* esc = (float*)(ws + off); off += (size_t)ETOT * H1C * 4;     // 13.6 MB
  float* alS = (float*)(ws + off); off += (size_t)NNODES * H1C * 4;
  float* alD = (float*)(ws + off); off += (size_t)NNODES * H1C * 4;
  unsigned* menc = (unsigned*)(ws + off); off += (size_t)NNODES * H1C * 4;
  float* den = (float*)(ws + off); off += (size_t)NNODES * H1C * 4;

  float* mu_out = (float*)d_out;
  float* ls_out = mu_out + (size_t)NNODES * CHID;

  // conv1: 128 -> 4 heads x 32, concat, then ELU -> h1
  run_gat(x, W1, a1s, a1d, b1, H1C, /*concat=*/true,
          srcE, dstE, xl, acc, esc, alS, alD, menc, den, h1, stream);
  // conv_mu: 128 -> 2 heads x 32, head-mean -> mu
  run_gat(h1, Wmu, amus, amud, bmu, H2C, /*concat=*/false,
          srcE, dstE, xl, acc, esc, alS, alD, menc, den, mu_out, stream);
  // conv_logstd: same shape -> logstd
  run_gat(h1, Wls, alss, alsd, bls, H2C, /*concat=*/false,
          srcE, dstE, xl, acc, esc, alS, alD, menc, den, ls_out, stream);
}
